// TokenMerge_50251117363356
// MI455X (gfx1250) — compile-verified, matrix operand order checked by reading, not run on hardware
//
#include <hip/hip_runtime.h>
#include <hip/hip_bf16.h>
#include <math.h>

// ---------------------------------------------------------------------------
// ToMe token merge for MI455X (gfx1250, wave32).
// B=8, T=4096 (2048 src even / 2048 dst odd), C=512, R=1024.
// Strategy: fused bf16-WMMA GEMM + row argmax (scores never hit memory),
// bitonic top-R selection in LDS, then bandwidth-bound gather/scatter merge.
// ---------------------------------------------------------------------------

typedef __bf16 bf16_t;
typedef bf16_t v16bf __attribute__((ext_vector_type(16)));
typedef bf16_t v8bf  __attribute__((ext_vector_type(8)));
typedef float  v8f   __attribute__((ext_vector_type(8)));

#define TB      8
#define TT      4096
#define TS      2048     // src / dst tokens per batch
#define TC      512
#define RNUM    1024
#define MROWS   3072     // merged rows per batch = (TS - R) + TS

// ---------------------------------------------------------------------------
// 1) Normalize rows of k, split even->A, odd->B, emit bf16.
//    One wave per row of 512 floats; lane holds 16 contiguous floats.
// ---------------------------------------------------------------------------
__global__ __launch_bounds__(256) void tm_normalize(
    const float* __restrict__ kin, bf16_t* __restrict__ An, bf16_t* __restrict__ Bn)
{
    int wave = threadIdx.x >> 5;
    int lane = threadIdx.x & 31;
    long row = (long)blockIdx.x * 8 + wave;      // 0 .. 8*4096-1
    int b = (int)(row >> 12);
    int t = (int)(row & 4095);

    const float* src = kin + row * TC + lane * 16;
    float v[16];
    float ss = 0.f;
#pragma unroll
    for (int i = 0; i < 16; ++i) { v[i] = src[i]; ss += v[i] * v[i]; }
#pragma unroll
    for (int m = 16; m >= 1; m >>= 1) ss += __shfl_xor(ss, m, 32);
    float scale = 1.0f / (sqrtf(ss) + 1e-12f);

    bf16_t* dst = ((t & 1) ? Bn : An) + ((size_t)b * TS + (t >> 1)) * TC + lane * 16;
#pragma unroll
    for (int i = 0; i < 16; ++i) dst[i] = (bf16_t)(v[i] * scale);
}

// ---------------------------------------------------------------------------
// 2) Fused scores GEMM + row max/argmax.
//    4 waves per block; each wave owns one 16-row A strip (A frags for all
//    K=512 held in 128 VGPRs) and sweeps 128 column tiles of B with
//    v_wmma_f32_16x16x32_bf16, tracking per-lane running (max, idx).
// ---------------------------------------------------------------------------
__global__ __launch_bounds__(128) void tm_scores_argmax(
    const bf16_t* __restrict__ An, const bf16_t* __restrict__ Bn,
    float* __restrict__ node_max, int* __restrict__ node_idx)
{
    int wave   = threadIdx.x >> 5;
    int lane   = threadIdx.x & 31;
    int gstrip = blockIdx.x * 4 + wave;          // 0 .. 8*128-1
    int batch  = gstrip >> 7;
    int strip  = (gstrip & 127) * 16;            // src-row base within batch
    int half   = lane >> 4;                      // which 16-lane half
    int l15    = lane & 15;

    const bf16_t* A = An + (size_t)batch * TS * TC;
    const bf16_t* B = Bn + (size_t)batch * TS * TC;

    // Preload all 16 K-chunk A fragments for this strip (register resident).
    // 16-bit A layout: lane<16 holds row l15, K = {k0..k0+7, k0+16..k0+23};
    // lane>=16 same row set, K shifted by 8.
    v16bf afrag[16];
    {
        const bf16_t* arow = A + (size_t)(strip + l15) * TC;
#pragma unroll
        for (int kc = 0; kc < 16; ++kc) {
            int k0 = kc * 32;
            v8bf lo = *(const v8bf*)(arow + k0 + half * 8);
            v8bf hi = *(const v8bf*)(arow + k0 + 16 + half * 8);
#pragma unroll
            for (int i = 0; i < 8; ++i) { afrag[kc][i] = lo[i]; afrag[kc][8 + i] = hi[i]; }
        }
    }

    float maxv[8];
    int   maxi[8];
#pragma unroll
    for (int r = 0; r < 8; ++r) { maxv[r] = -__builtin_inff(); maxi[r] = 0; }

    for (int tile = 0; tile < 128; ++tile) {
        const bf16_t* brow = B + (size_t)(tile * 16 + l15) * TC;  // col l15 of tile, K-contig
        v8f c = {};
#pragma unroll
        for (int kc = 0; kc < 16; ++kc) {
            int k0 = kc * 32;
            v8bf lo = *(const v8bf*)(brow + k0 + half * 8);
            v8bf hi = *(const v8bf*)(brow + k0 + 16 + half * 8);
            v16bf bfrag;
#pragma unroll
            for (int i = 0; i < 8; ++i) { bfrag[i] = lo[i]; bfrag[8 + i] = hi[i]; }
            c = __builtin_amdgcn_wmma_f32_16x16x32_bf16(
                    false, afrag[kc], false, bfrag, (short)0, c, false, false);
        }
        // C layout: VGPR r, this lane -> row strip + r + 8*half, col tile*16 + l15
        int col = tile * 16 + l15;
#pragma unroll
        for (int r = 0; r < 8; ++r) {
            if (c[r] > maxv[r]) { maxv[r] = c[r]; maxi[r] = col; }
        }
    }

    // Reduce (max, argmax) across the 16 lanes of each half (ties -> lower idx).
#pragma unroll
    for (int m = 1; m < 16; m <<= 1) {
#pragma unroll
        for (int r = 0; r < 8; ++r) {
            float ov = __shfl_xor(maxv[r], m, 32);
            int   oi = __shfl_xor(maxi[r], m, 32);
            if (ov > maxv[r] || (ov == maxv[r] && oi < maxi[r])) { maxv[r] = ov; maxi[r] = oi; }
        }
    }
    if (l15 == 0) {
#pragma unroll
        for (int r = 0; r < 8; ++r) {
            int row = strip + r + 8 * half;
            node_max[batch * TS + row] = maxv[r];
            node_idx[batch * TS + row] = maxi[r];
        }
    }
}

// ---------------------------------------------------------------------------
// 3) Per-batch selection: bitonic sort of 2048 (value,idx) pairs descending
//    (ties ascending idx == stable argsort(-v)), then sort the unmerged 1024
//    indices ascending. PROTECT_CLS: row 0 forced to -inf.
// ---------------------------------------------------------------------------
__global__ __launch_bounds__(1024) void tm_select(
    const float* __restrict__ node_max,
    int* __restrict__ src_idx, int* __restrict__ unm_sorted)
{
    __shared__ float sv[TS];
    __shared__ int   si[TS];
    int b   = blockIdx.x;
    int tid = threadIdx.x;

    for (int i = tid; i < TS; i += 1024) {
        float v = node_max[b * TS + i];
        if (i == 0) v = -__builtin_inff();       // PROTECT_CLS
        sv[i] = v; si[i] = i;
    }
    __syncthreads();

    for (int k = 2; k <= TS; k <<= 1) {
        for (int j = k >> 1; j > 0; j >>= 1) {
            for (int t = tid; t < TS; t += 1024) {
                int ixj = t ^ j;
                if (ixj > t) {
                    bool up = ((t & k) == 0);
                    float v0 = sv[t], v1 = sv[ixj];
                    int   i0 = si[t], i1 = si[ixj];
                    bool before = (v0 > v1) || (v0 == v1 && i0 < i1);
                    if (before != up) { sv[t] = v1; sv[ixj] = v0; si[t] = i1; si[ixj] = i0; }
                }
            }
            __syncthreads();
        }
    }

    if (tid < RNUM) src_idx[b * RNUM + tid] = si[tid];   // edge_order[:R], in order
    __syncthreads();

    // ascending index sort of the unmerged tail si[1024..2047]
    for (int k = 2; k <= RNUM; k <<= 1) {
        for (int j = k >> 1; j > 0; j >>= 1) {
            int t = tid;
            int ixj = t ^ j;
            if (t < RNUM && ixj > t) {
                int a0 = si[RNUM + t], a1 = si[RNUM + ixj];
                bool up = ((t & k) == 0);
                if ((a0 < a1) != up) { si[RNUM + t] = a1; si[RNUM + ixj] = a0; }
            }
            __syncthreads();
        }
    }
    if (tid < RNUM) unm_sorted[b * RNUM + tid] = si[RNUM + tid];
}

// ---------------------------------------------------------------------------
// 4a) merged[:, :1024] = gather(src_tok, unm_sorted); merged[:, 1024:2048] = dst_tok[:1024]
// ---------------------------------------------------------------------------
__global__ __launch_bounds__(256) void tm_merge_copy(
    const float* __restrict__ x, const int* __restrict__ unm_sorted,
    float* __restrict__ merged)
{
    int row = blockIdx.x;                 // 0 .. 8*2048-1
    int b = row >> 11;
    int r = row & 2047;
    const float* src;
    if (r < RNUM) {
        int s = unm_sorted[b * RNUM + r];
        src = x + ((size_t)b * TT + 2 * s) * TC;         // even (src) token
    } else {
        int d = r - RNUM;
        src = x + ((size_t)b * TT + 2 * d + 1) * TC;     // odd (dst) token
    }
    float* dst = merged + ((size_t)b * MROWS + r) * TC;
#pragma unroll
    for (int i = 0; i < 2; ++i)
        dst[threadIdx.x + 256 * i] = src[threadIdx.x + 256 * i];
}

// ---------------------------------------------------------------------------
// 4b) scatter-add merged src tokens into their dst rows
// ---------------------------------------------------------------------------
__global__ __launch_bounds__(256) void tm_scatter_add(
    const float* __restrict__ x, const int* __restrict__ src_idx,
    const int* __restrict__ node_idx, float* __restrict__ merged)
{
    int j = blockIdx.x;                   // 0 .. 8*1024-1
    int b = j >> 10;
    int e = j & 1023;
    int s = src_idx[b * RNUM + e];
    int d = node_idx[b * TS + s];
    const float* src = x + ((size_t)b * TT + 2 * s) * TC;
    float* dst = merged + ((size_t)b * MROWS + RNUM + d) * TC;
#pragma unroll
    for (int i = 0; i < 2; ++i)
        atomicAdd(&dst[threadIdx.x + 256 * i], src[threadIdx.x + 256 * i]);
}

// ---------------------------------------------------------------------------
// 5) source_index: every position written exactly once (odd -> dst slot,
//    even -> either unm rank or merged dst slot), so no -1 init needed.
// ---------------------------------------------------------------------------
__global__ __launch_bounds__(1024) void tm_source_index(
    const int* __restrict__ src_idx, const int* __restrict__ unm_sorted,
    const int* __restrict__ node_idx, int* __restrict__ si_out)
{
    int b = blockIdx.x;
    int tid = threadIdx.x;
    int* si = si_out + b * TT;
    for (int d = tid; d < TS; d += 1024)
        si[2 * d + 1] = RNUM + d;
    {
        int e = tid;                       // 0..1023
        int s = src_idx[b * RNUM + e];
        si[2 * s] = RNUM + node_idx[b * TS + s];
        int u = unm_sorted[b * RNUM + e];
        si[2 * u] = e;
    }
}

// ---------------------------------------------------------------------------
extern "C" void kernel_launch(void* const* d_in, const int* in_sizes, int n_in,
                              void* d_out, int out_size, void* d_ws, size_t ws_size,
                              hipStream_t stream) {
    const float* x = (const float*)d_in[0];
    const float* k = (const float*)d_in[1];

    char* ws = (char*)d_ws;
    const size_t panel = (size_t)TB * TS * TC * sizeof(bf16_t);   // 16 MiB
    bf16_t* An        = (bf16_t*)(ws);
    bf16_t* Bn        = (bf16_t*)(ws + panel);
    float*  node_max  = (float*)(ws + 2 * panel);
    int*    node_idx  = (int*)  (ws + 2 * panel + (size_t)TB * TS * 4);
    int*    src_idx   = (int*)  (ws + 2 * panel + (size_t)TB * TS * 8);
    int*    unm_sort  = (int*)  (ws + 2 * panel + (size_t)TB * TS * 8 + (size_t)TB * RNUM * 4);

    float* merged = (float*)d_out;                              // 8*3072*512 floats
    int*   si_out = (int*)(merged + (size_t)TB * MROWS * TC);   // 8*4096 int32 (bit layout)

    tm_normalize   <<<TB * TT / 8,   256, 0, stream>>>(k, An, Bn);
    tm_scores_argmax<<<TB * 128 / 4, 128, 0, stream>>>(An, Bn, node_max, node_idx);
    tm_select      <<<TB,           1024, 0, stream>>>(node_max, src_idx, unm_sort);
    tm_merge_copy  <<<TB * TS,       256, 0, stream>>>(x, unm_sort, merged);
    tm_scatter_add <<<TB * RNUM,     256, 0, stream>>>(x, src_idx, node_idx, merged);
    tm_source_index<<<TB,           1024, 0, stream>>>(src_idx, unm_sort, node_idx, si_out);
}
